// FisherBTreeNetwork_2542620639788
// MI455X (gfx1250) — compile-verified, matrix operand order checked by reading, not run on hardware
//
#include <hip/hip_runtime.h>
#include <hip/hip_bf16.h>
#include <stdint.h>

// ---------------------------------------------------------------------------
// MLP forward for MI455X (gfx1250, wave32):
//   h     = relu(x @ W1 + b1)
//   feats = relu(h @ W2 + b2)
//   conf  = softmax(feats @ Wc + bc)
// bf16 WMMA (v_wmma_f32_16x16x32_bf16). Block tile 256x128x32, 8 waves with
// 64x64 wave tiles (16 WMMA per k-chunk, 1.0 ds_load_b128 per WMMA).
// Double-buffered LDS fed by CDNA5 async global->LDS copies (ASYNCcnt).
// Fused bias+relu epilogue, bf16 intermediates.
// ---------------------------------------------------------------------------

typedef __attribute__((ext_vector_type(16))) __bf16 v16bf;
typedef __attribute__((ext_vector_type(8)))  __bf16 v8bf;
typedef __attribute__((ext_vector_type(8)))  float  v8f;

#define BM 256
#define BN 128
#define BK 32

static const long B_SZ  = 8192;
static const long D_IN  = 4096;
static const long D_H   = 4096;
static const long D_OUT = 1000;
static const long D_OUTP = 1024;   // padded classifier width (multiple of 16)

// --- CDNA5 async global->LDS copy (tracked by ASYNCcnt), inline asm --------
__device__ __forceinline__ void async_copy_b128(unsigned lds_off,
                                                const void* gptr) {
  asm volatile("global_load_async_to_lds_b128 %0, %1, off"
               :: "v"(lds_off), "v"(gptr)
               : "memory");
}
__device__ __forceinline__ void wait_async0() {
  asm volatile("s_wait_asynccnt 0x0" ::: "memory");
}
__device__ __forceinline__ unsigned lds_off_of(const void* p) {
  return (unsigned)(uintptr_t)p;   // flat LDS addr: low 32 bits = LDS offset
}

// ---------------- elementwise f32 -> bf16 (8 elems / thread) ---------------
__global__ __launch_bounds__(256) void k_cvt_bf16(const float* __restrict__ in,
                                                  __bf16* __restrict__ out,
                                                  long n) {
  long i = ((long)blockIdx.x * blockDim.x + threadIdx.x) * 8;
  if (i + 8 <= n) {
    float4 f0 = *(const float4*)(in + i);
    float4 f1 = *(const float4*)(in + i + 4);
    v8bf o = {(__bf16)f0.x, (__bf16)f0.y, (__bf16)f0.z, (__bf16)f0.w,
              (__bf16)f1.x, (__bf16)f1.y, (__bf16)f1.z, (__bf16)f1.w};
    *(v8bf*)(out + i) = o;
  } else {
    for (long j = i; j < n; ++j) out[j] = (__bf16)in[j];
  }
}

// ------------- weight transpose+convert: W[K][N] f32 -> Wt[Nout][K] bf16 ----
// Rows n >= N are zero-padded (used for the 1000 -> 1024 classifier pad).
__global__ __launch_bounds__(256) void k_cvtT(const float* __restrict__ W,
                                              __bf16* __restrict__ Wt,
                                              int K, int N) {
  int k = blockIdx.x * blockDim.x + threadIdx.x;   // contiguous in output
  int n = blockIdx.y;
  if (k < K)
    Wt[(long)n * K + k] = (n < N) ? (__bf16)W[(long)k * N + n] : (__bf16)0.0f;
}

// ---------------- tiled bf16 WMMA GEMM + fused epilogue --------------------
// A  : [M][K] bf16 row-major
// Bt : [N][K] bf16 (i.e. B transposed so K is contiguous for both operands)
// RELU_BF16OUT=1 -> out = relu(acc+bias) as bf16 [M][N]
// RELU_BF16OUT=0 -> out = acc+bias as f32 [M][N] (bias 0 for col >= nbias)
template <bool RELU_BF16OUT>
__global__ __launch_bounds__(256) void k_gemm_bf16(
    const __bf16* __restrict__ A, const __bf16* __restrict__ Bt,
    const float* __restrict__ bias, int nbias,
    __bf16* __restrict__ outBf, float* __restrict__ outF,
    int M, int N, int K) {
  __shared__ __bf16 sA[2][BM * BK];   // [buf][row][k]  2 x 16 KB
  __shared__ __bf16 sB[2][BN * BK];   // [buf][col][k]  2 x  8 KB

  const int tid  = threadIdx.x;
  const int lane = tid & 31;
  const int wave = tid >> 5;       // 8 waves
  const int wm   = wave & 3;       // 4 waves along M -> 64 rows each
  const int wn   = wave >> 2;      // 2 waves along N -> 64 cols each
  const long bm0 = (long)blockIdx.x * BM;
  const long bn0 = (long)blockIdx.y * BN;

  // cooperative tile staging: 4 threads per 32-wide k row, 8 bf16 (16B) each.
  // A: rows r0 + 64*s (s=0..3); B: rows r0 + 64*s (s=0..1).
  const int q  = tid & 3;
  const int r0 = tid >> 2;         // 0..63
  const __bf16* gA[4];
  const __bf16* gB[2];
  unsigned oA[2][4], oB[2][2];
#pragma unroll
  for (int s = 0; s < 4; ++s) gA[s] = A + (bm0 + r0 + 64 * s) * K + q * 8;
#pragma unroll
  for (int s = 0; s < 2; ++s) gB[s] = Bt + (bn0 + r0 + 64 * s) * K + q * 8;
#pragma unroll
  for (int b = 0; b < 2; ++b) {
#pragma unroll
    for (int s = 0; s < 4; ++s)
      oA[b][s] = lds_off_of(&sA[b][(r0 + 64 * s) * BK + q * 8]);
#pragma unroll
    for (int s = 0; s < 2; ++s)
      oB[b][s] = lds_off_of(&sB[b][(r0 + 64 * s) * BK + q * 8]);
  }

  // WMMA A/B fragment addressing (16-bit 16x32 layout):
  //   lane holds matrix row (A) / col (B)  = lane & 15
  //   k coverage: base..base+7 and base+16..base+23, base = lane<16 ? 0 : 8
  const int fr    = lane & 15;
  const int kbase = (lane < 16) ? 0 : 8;

  v8f acc[4][4];
#pragma unroll
  for (int i = 0; i < 4; ++i)
#pragma unroll
    for (int j = 0; j < 4; ++j) acc[i][j] = (v8f)0.0f;

  // stage chunk 0 into buffer 0
#pragma unroll
  for (int s = 0; s < 4; ++s) async_copy_b128(oA[0][s], gA[s]);
#pragma unroll
  for (int s = 0; s < 2; ++s) async_copy_b128(oB[0][s], gB[s]);
  wait_async0();
  __syncthreads();

  int buf = 0;
  for (int k0 = 0; k0 < K; k0 += BK) {
    // prefetch chunk k0+BK into the other buffer (hidden behind WMMAs)
    if (k0 + BK < K) {
      const int nb = buf ^ 1;
#pragma unroll
      for (int s = 0; s < 4; ++s) async_copy_b128(oA[nb][s], gA[s] + k0 + BK);
#pragma unroll
      for (int s = 0; s < 2; ++s) async_copy_b128(oB[nb][s], gB[s] + k0 + BK);
    }

    v16bf aF[4], bF[4];
#pragma unroll
    for (int i = 0; i < 4; ++i) {
      const __bf16* p = &sA[buf][(wm * 64 + i * 16 + fr) * BK + kbase];
      v8bf lo = *(const v8bf*)p;           // k = kbase .. kbase+7
      v8bf hi = *(const v8bf*)(p + 16);    // k = kbase+16 .. kbase+23
      aF[i] = __builtin_shufflevector(lo, hi, 0, 1, 2, 3, 4, 5, 6, 7, 8, 9, 10,
                                      11, 12, 13, 14, 15);
    }
#pragma unroll
    for (int j = 0; j < 4; ++j) {
      const __bf16* p = &sB[buf][(wn * 64 + j * 16 + fr) * BK + kbase];
      v8bf lo = *(const v8bf*)p;
      v8bf hi = *(const v8bf*)(p + 16);
      bF[j] = __builtin_shufflevector(lo, hi, 0, 1, 2, 3, 4, 5, 6, 7, 8, 9, 10,
                                      11, 12, 13, 14, 15);
    }
#pragma unroll
    for (int i = 0; i < 4; ++i)
#pragma unroll
      for (int j = 0; j < 4; ++j)
        acc[i][j] = __builtin_amdgcn_wmma_f32_16x16x32_bf16(
            false, aF[i], false, bF[j], (short)0, acc[i][j], false, false);

    wait_async0();     // our async writes into buf^1 are done
    __syncthreads();   // everyone done reading buf / writing buf^1
    buf ^= 1;
  }

  // Epilogue. C/D layout: VGPR e, lanes 0-15 -> row e, lanes 16-31 -> row e+8;
  // col = lane & 15.
  const int rowHalf = (lane >> 4) << 3;
#pragma unroll
  for (int i = 0; i < 4; ++i) {
#pragma unroll
    for (int j = 0; j < 4; ++j) {
      const long col = bn0 + wn * 64 + j * 16 + fr;
      const float bv = (col < nbias) ? bias[col] : 0.0f;
#pragma unroll
      for (int e = 0; e < 8; ++e) {
        const long row = bm0 + wm * 64 + i * 16 + rowHalf + e;
        float v = acc[i][j][e] + bv;
        if (RELU_BF16OUT) {
          v = v > 0.0f ? v : 0.0f;
          outBf[row * N + col] = (__bf16)v;
        } else {
          outF[row * N + col] = v;
        }
      }
    }
  }
}

// ---------------- row softmax over the first ncols of [ldl]-wide rows ------
__global__ __launch_bounds__(256) void k_softmax(const float* __restrict__ lg,
                                                 float* __restrict__ out,
                                                 int ncols, int ldl) {
  __shared__ float red[256];
  const long row = blockIdx.x;
  const float* lp = lg + row * ldl;
  float m = -INFINITY;
  for (int c = threadIdx.x; c < ncols; c += 256) m = fmaxf(m, lp[c]);
  red[threadIdx.x] = m;
  __syncthreads();
  for (int s = 128; s > 0; s >>= 1) {
    if (threadIdx.x < s) red[threadIdx.x] = fmaxf(red[threadIdx.x], red[threadIdx.x + s]);
    __syncthreads();
  }
  m = red[0];
  __syncthreads();
  float sum = 0.0f;
  for (int c = threadIdx.x; c < ncols; c += 256) sum += __expf(lp[c] - m);
  red[threadIdx.x] = sum;
  __syncthreads();
  for (int s = 128; s > 0; s >>= 1) {
    if (threadIdx.x < s) red[threadIdx.x] += red[threadIdx.x + s];
    __syncthreads();
  }
  const float inv = 1.0f / red[0];
  for (int c = threadIdx.x; c < ncols; c += 256)
    out[row * ncols + c] = __expf(lp[c] - m) * inv;
}

// ---------------------------------------------------------------------------
extern "C" void kernel_launch(void* const* d_in, const int* in_sizes, int n_in,
                              void* d_out, int out_size, void* d_ws,
                              size_t ws_size, hipStream_t stream) {
  (void)in_sizes; (void)n_in; (void)out_size; (void)ws_size;
  const float* x  = (const float*)d_in[0];
  const float* W1 = (const float*)d_in[1];
  const float* b1 = (const float*)d_in[2];
  const float* W2 = (const float*)d_in[3];
  const float* b2 = (const float*)d_in[4];
  const float* Wc = (const float*)d_in[5];
  const float* bc = (const float*)d_in[6];
  float* out = (float*)d_out;

  // workspace layout (~296 MB)
  char* ws = (char*)d_ws;
  __bf16* xb  = (__bf16*)ws;  ws += B_SZ * D_IN * 2;
  __bf16* W1t = (__bf16*)ws;  ws += D_IN * D_H * 2;
  __bf16* W2t = (__bf16*)ws;  ws += D_H * D_H * 2;
  __bf16* Wct = (__bf16*)ws;  ws += D_OUTP * D_H * 2;
  __bf16* hb  = (__bf16*)ws;  ws += B_SZ * D_H * 2;
  __bf16* fb  = (__bf16*)ws;  ws += B_SZ * D_H * 2;
  float*  lg  = (float*)ws;   ws += B_SZ * D_OUTP * 4;

  // 1) convert inputs / transpose weights to bf16
  {
    long n = B_SZ * D_IN;
    k_cvt_bf16<<<dim3((n / 8 + 255) / 256), dim3(256), 0, stream>>>(x, xb, n);
  }
  k_cvtT<<<dim3(D_IN / 256, D_H), dim3(256), 0, stream>>>(W1, W1t, D_IN, D_H);
  k_cvtT<<<dim3(D_H / 256, D_H), dim3(256), 0, stream>>>(W2, W2t, D_H, D_H);
  k_cvtT<<<dim3(D_H / 256, D_OUTP), dim3(256), 0, stream>>>(Wc, Wct, D_H, D_OUT);

  const dim3 blk(256);
  // 2) h = relu(x @ W1 + b1)   [8192 x 4096]
  k_gemm_bf16<true><<<dim3(B_SZ / BM, D_H / BN), blk, 0, stream>>>(
      xb, W1t, b1, (int)D_H, hb, nullptr, (int)B_SZ, (int)D_H, (int)D_IN);
  // 3) feats = relu(h @ W2 + b2)
  k_gemm_bf16<true><<<dim3(B_SZ / BM, D_H / BN), blk, 0, stream>>>(
      hb, W2t, b2, (int)D_H, fb, nullptr, (int)B_SZ, (int)D_H, (int)D_H);
  // 4) logits = feats @ Wc + bc  (padded to 1024 cols, bias 0 past 1000)
  k_gemm_bf16<false><<<dim3(B_SZ / BM, D_OUTP / BN), blk, 0, stream>>>(
      fb, Wct, bc, (int)D_OUT, nullptr, lg, (int)B_SZ, (int)D_OUTP, (int)D_H);
  // 5) softmax over the 1000 valid columns
  k_softmax<<<dim3(B_SZ), blk, 0, stream>>>(lg, out, (int)D_OUT, (int)D_OUTP);
}
